// MaskedBCE_77257871721093
// MI455X (gfx1250) — compile-verified, matrix operand order checked by reading, not run on hardware
//
#include <hip/hip_runtime.h>

// MI455X / gfx1250, wave32.
typedef __attribute__((ext_vector_type(2)))  float    v2f;
typedef __attribute__((ext_vector_type(4)))  float    v4f;
typedef __attribute__((ext_vector_type(8)))  float    v8f;
typedef __attribute__((ext_vector_type(16))) _Float16 v16h;

#define B_   64
#define T_   2048
#define D_   88
#define TOTAL4_ ((B_ * T_ * D_) / 4)   // 2,883,584 float4 elements
#define NBLK 2048
#define NTHR 256                        // 8 wave32s per block

// Stable BCE-with-logits: max(x,0) - x*t + log1p(exp(-|x|)).
// log1p(exp(-a)) for a>=0: e in (0,1], log(1+e) is well conditioned -> fast log.
__device__ __forceinline__ float bce1(float x, float t) {
  float ax = __builtin_fabsf(x);
  float e  = __expf(-ax);           // v_exp_f32 (trans, co-executes with VALU)
  float l  = __logf(1.0f + e);      // v_log_f32
  return fmaxf(x, 0.0f) - x * t + l;
}

// Wave32 all-lane sum via WMMA: A[16x4] holds the 32 lane partials
// (lane m<16 -> A[m][0], lane m+16 -> A[m][2]); B = ones => D[m][n] = rowsum[m].
// Lanes 0-15 see sum of rows 0-7 in d[0..7]; lanes 16-31 see rows 8-15.
// One xor-16 shuffle combines the halves -> total in every lane. Exact fp32.
__device__ __forceinline__ float wave_sum_wmma(float acc) {
  v8f c = {};
  v8f d;
#if __has_builtin(__builtin_amdgcn_wmma_f32_16x16x4_f32)
  v2f a; a.x = acc;  a.y = 0.0f;
  v2f b; b.x = 1.0f; b.y = 1.0f;
  d = __builtin_amdgcn_wmma_f32_16x16x4_f32(false, a, false, b, (short)0, c,
                                            false, false);
#else
  // Fallback (codegen-confirmed builtin): same reduction structure via f16 A/B.
  v16h a = {};
  a[0] = (_Float16)acc;
  v16h b;
#pragma unroll
  for (int k = 0; k < 16; ++k) b[k] = (_Float16)1.0f;
  d = __builtin_amdgcn_wmma_f32_16x16x32_f16(false, a, false, b, (short)0, c,
                                             false, false);
#endif
  float s = d[0] + d[1] + d[2] + d[3] + d[4] + d[5] + d[6] + d[7];
  s += __shfl_xor(s, 16, 32);
  return s;
}

// Kernel 1: invT[b] = 1 / sum_t mask[b,t]   (64 blocks, tiny)
__global__ void k_lengths(const float* __restrict__ mask, float* __restrict__ invT) {
  __shared__ float sm[NTHR];
  const int b = blockIdx.x, tid = threadIdx.x;
  float s = 0.0f;
  for (int t = tid; t < T_; t += NTHR) s += mask[b * T_ + t];
  sm[tid] = s;
  __syncthreads();
  for (int off = NTHR / 2; off > 0; off >>= 1) {
    if (tid < off) sm[tid] += sm[tid + off];
    __syncthreads();
  }
  if (tid == 0) invT[b] = 1.0f / sm[0];   // lengths >= 1, never div-by-zero
}

// Kernel 2: streaming masked BCE, one fp32 partial per block.
__global__ void k_bce(const v4f* __restrict__ x4, const v4f* __restrict__ t4,
                      const float* __restrict__ mask,
                      const float* __restrict__ invT,
                      float* __restrict__ blockSums) {
  float acc = 0.0f;
  const int stride = gridDim.x * blockDim.x;
  for (int i = blockIdx.x * blockDim.x + threadIdx.x; i < TOTAL4_; i += stride) {
    const int e  = i << 2;        // element index (fits int32)
    const int bt = e / D_;        // row index; D_=88 -> mul+shift
    const float m = mask[bt];     // L2-resident, checked BEFORE big loads
    if (m != 0.0f) {              // prefix mask -> warp-coherent skip (~50% HBM saved)
      const int   b = bt >> 11;   // / T_ (2048)
      const float w = m * invT[b];
      v4f x = __builtin_nontemporal_load(&x4[i]);   // single-pass: NT hint
      v4f t = __builtin_nontemporal_load(&t4[i]);
      float s4 = bce1(x.x, t.x) + bce1(x.y, t.y) +
                 bce1(x.z, t.z) + bce1(x.w, t.w);
      acc += w * s4;
    }
  }
  // Full EXEC here (all lanes exit the loop) -> WMMA legal.
  const float wsum = wave_sum_wmma(acc);
  __shared__ float sm[NTHR / 32];
  if ((threadIdx.x & 31) == 0) sm[threadIdx.x >> 5] = wsum;
  __syncthreads();
  if (threadIdx.x == 0) {
    float tot = 0.0f;
#pragma unroll
    for (int w = 0; w < NTHR / 32; ++w) tot += sm[w];
    blockSums[blockIdx.x] = tot;   // no atomics: deterministic
  }
}

// Kernel 3: fixed-order tree reduction of block partials -> scalar.
__global__ void k_final(const float* __restrict__ blockSums, int n,
                        float* __restrict__ out) {
  __shared__ float sm[NTHR];
  const int tid = threadIdx.x;
  float s = 0.0f;
  for (int i = tid; i < n; i += NTHR) s += blockSums[i];
  sm[tid] = s;
  __syncthreads();
  for (int off = NTHR / 2; off > 0; off >>= 1) {
    if (tid < off) sm[tid] += sm[tid + off];
    __syncthreads();
  }
  if (tid == 0) out[0] = sm[0];
}

extern "C" void kernel_launch(void* const* d_in, const int* in_sizes, int n_in,
                              void* d_out, int out_size, void* d_ws, size_t ws_size,
                              hipStream_t stream) {
  (void)in_sizes; (void)n_in; (void)out_size; (void)ws_size;
  const float* output = (const float*)d_in[0];   // [B,T,D] logits
  const float* target = (const float*)d_in[1];   // [B,T,D] targets
  const float* mask   = (const float*)d_in[2];   // [B,T]

  float* invT      = (float*)d_ws;       // 64 floats
  float* blockSums = invT + B_;          // NBLK floats  (total 8448 B of ws)
  float* out       = (float*)d_out;

  k_lengths<<<B_, NTHR, 0, stream>>>(mask, invT);
  k_bce<<<NBLK, NTHR, 0, stream>>>((const v4f*)output, (const v4f*)target,
                                   mask, invT, blockSums);
  k_final<<<1, NTHR, 0, stream>>>(blockSums, NBLK, out);
}
